// LSTM_CRF_26044681683473
// MI455X (gfx1250) — matrix operand order repaired and checked by reading, not compiled, not run on hardware
//
#include <hip/hip_runtime.h>

// ---------------------------------------------------------------------------
// BiLSTM + CRF for MI455X (gfx1250, wave32, WMMA, async-LDS, TDM).
// GEMMs: v_wmma_f32_16x16x32_bf16 (bf16 in, f32 accumulate).
// xproj: double-buffered GLOBAL_LOAD_ASYNC_TO_LDS staging (ASYNCcnt).
// clf:   Tensor Data Mover tile loads (TENSORcnt).
// ---------------------------------------------------------------------------

#define Bsz   64
#define Tsz   512
#define Dsz   512
#define Hsz   256
#define G4H   1024   // 4*H
#define Kst   32     // CRF states

typedef __attribute__((ext_vector_type(16))) __bf16     v16bf;
typedef __attribute__((ext_vector_type(8)))  float      v8f;
typedef __attribute__((ext_vector_type(4)))  unsigned   u32x4;
typedef __attribute__((ext_vector_type(8)))  unsigned   u32x8;
typedef __attribute__((ext_vector_type(4)))  int        i32x4;
typedef __attribute__((ext_vector_type(8)))  int        i32x8;

#if __has_builtin(__builtin_amdgcn_tensor_load_to_lds) && __has_builtin(__builtin_amdgcn_s_wait_tensorcnt)
#define HAVE_TDM 1
#endif

__device__ __forceinline__ unsigned short f2bf(float f) {
    unsigned u = __float_as_uint(f);
    unsigned r = u + 0x7FFFu + ((u >> 16) & 1u);   // round-to-nearest-even
    return (unsigned short)(r >> 16);
}
__device__ __forceinline__ float bf2f(unsigned short h) {
    return __uint_as_float(((unsigned)h) << 16);
}
__device__ __forceinline__ float sigm(float x) {
    return 1.0f / (1.0f + __expf(-x));
}
__device__ __forceinline__ v16bf frag2(u32x4 a, u32x4 b) {   // two 16B K-runs (A layout)
    u32x8 t; t.lo = a; t.hi = b;
    return __builtin_bit_cast(v16bf, t);
}
__device__ __forceinline__ v16bf frag1(const unsigned short* p) { // two 16B K-runs (B layout)
    u32x8 t; t.lo = *(const u32x4*)p; t.hi = *(const u32x4*)(p + 16);
    return __builtin_bit_cast(v16bf, t);
}
#define WMMA_BF16(A, Bm, C) \
    __builtin_amdgcn_wmma_f32_16x16x32_bf16(false, (A), false, (Bm), (short)0, (C), false, false)

// ---- CDNA5 async/TDM helpers ----------------------------------------------
__device__ __forceinline__ unsigned lds_byte_addr(const unsigned short* p) {
    typedef __attribute__((address_space(3))) const unsigned short lds_cu16;
    return (unsigned)(unsigned long long)(lds_cu16*)p;
}
// One-lane-granular async copy: 16B from global to LDS, tracked by ASYNCcnt.
__device__ __forceinline__ void async_ld_b128(const unsigned short* lds_dst,
                                              const unsigned short* gsrc) {
    asm volatile("global_load_async_to_lds_b128 %0, %1, off"
                 :: "v"(lds_byte_addr(lds_dst)), "v"(gsrc)
                 : "memory");
}
__device__ __forceinline__ void wait_async0() {
    asm volatile("s_wait_asynccnt 0x0" ::: "memory");
}
#ifdef HAVE_TDM
// 2D TDM tile load: D# groups per CDNA5 ISA ch.8 (data_size=2B, type=2).
__device__ __forceinline__ void tdm_load_2d(const unsigned short* gaddr, unsigned ldsaddr,
                                            unsigned tdim0, unsigned tdim1,
                                            unsigned tile0, unsigned tile1,
                                            unsigned stride0) {
    unsigned long long ga = (unsigned long long)gaddr;
    u32x4 g0;
    g0.x = 1u;                                            // count=1, user descriptor
    g0.y = ldsaddr;                                       // lds_addr [63:32]
    g0.z = (unsigned)(ga & 0xFFFFFFFFu);                  // global_addr[31:0]
    g0.w = (unsigned)((ga >> 32) & 0x1FFFFFFu) | (2u << 30); // global_addr[56:32] | type=2
    i32x8 g1;
    g1.s0 = (int)(1u << 16);                              // wg_mask=0, data_size=1 (2 bytes)
    g1.s1 = (int)((tdim0 & 0xFFFFu) << 16);               // tensor_dim0[15:0]
    g1.s2 = (int)((tdim0 >> 16) | ((tdim1 & 0xFFFFu) << 16)); // dim0[31:16] | dim1[15:0]
    g1.s3 = (int)((tdim1 >> 16) | (tile0 << 16));         // dim1[31:16] | tile_dim0
    g1.s4 = (int)(tile1 & 0xFFFFu);                       // tile_dim1, tile_dim2=0
    g1.s5 = (int)stride0;                                 // tensor_dim0_stride[31:0]
    g1.s6 = 0; g1.s7 = 0;
    i32x4 z4 = (i32x4)0;
    i32x8 z8 = (i32x8)0;
    // 6-arg toolchain variant: (g0, g1, g2, g3, g4, cpol)
    __builtin_amdgcn_tensor_load_to_lds(g0, g1, z4, z4, z8, 0);
}
#endif

// ---------------------------------------------------------------------------
// f32 -> bf16 conversion (grid-stride)
// ---------------------------------------------------------------------------
__global__ void cvt_bf16_kernel(const float* __restrict__ in,
                                unsigned short* __restrict__ out, int n) {
    int i = blockIdx.x * blockDim.x + threadIdx.x;
    int stride = gridDim.x * blockDim.x;
    for (; i < n; i += stride) out[i] = f2bf(in[i]);
}

// ---------------------------------------------------------------------------
// xp = x @ Wih^T + b.  M = 32768, K = 512, N = 1024.  grid = (256, 8, 2).
// Block 256 thr (8 waves): WG tile 128x128, wave tile 64x32.
// A/B tiles double-buffered in LDS via GLOBAL_LOAD_ASYNC_TO_LDS_B128.
// ---------------------------------------------------------------------------
__global__ void __launch_bounds__(256)
xproj_kernel(const unsigned short* __restrict__ xbf,
             const unsigned short* __restrict__ wihf,
             const unsigned short* __restrict__ wihb,
             const float* __restrict__ biasf, const float* __restrict__ biasb,
             float* __restrict__ xpf, float* __restrict__ xpb) {
    const int dir = blockIdx.z;
    const unsigned short* wbf = dir ? wihb : wihf;
    const float* bias         = dir ? biasb : biasf;
    float* out                = dir ? xpb : xpf;

    __shared__ unsigned short Abuf[2][128 * 32];   // 2 x 8 KB
    __shared__ unsigned short Bbuf[2][128 * 32];   // 2 x 8 KB

    const int tid = threadIdx.x;
    const int w  = tid >> 5;
    const int l  = tid & 31;
    const int lr = l & 15, lh = l >> 4;
    const int wm = w & 1, wn = w >> 1;
    const int rowBase0 = blockIdx.x * 128;
    const int colBase0 = blockIdx.y * 128;

    // staging map: each thread copies one 16B segment per tile per K-step
    const int srow = tid >> 1, sseg = (tid & 1) * 8;
    const unsigned short* gA = xbf + (size_t)(rowBase0 + srow) * Dsz + sseg;
    const unsigned short* gB = wbf + (size_t)(colBase0 + srow) * Dsz + sseg;

    async_ld_b128(&Abuf[0][srow * 32 + sseg], gA);
    async_ld_b128(&Bbuf[0][srow * 32 + sseg], gB);
    wait_async0();
    __syncthreads();

    v8f acc[4][2];
    #pragma unroll
    for (int m = 0; m < 4; ++m) { acc[m][0] = (v8f)0.0f; acc[m][1] = (v8f)0.0f; }

    for (int kc = 0; kc < 16; ++kc) {
        const int cur = kc & 1;
        if (kc + 1 < 16) {                         // issue next-tile async copies
            async_ld_b128(&Abuf[cur ^ 1][srow * 32 + sseg], gA + (kc + 1) * 32);
            async_ld_b128(&Bbuf[cur ^ 1][srow * 32 + sseg], gB + (kc + 1) * 32);
        }
        v16bf b0 = frag1(&Bbuf[cur][(wn * 32 + lr)      * 32 + lh * 8]);
        v16bf b1 = frag1(&Bbuf[cur][(wn * 32 + 16 + lr) * 32 + lh * 8]);
        #pragma unroll
        for (int m = 0; m < 4; ++m) {
            const unsigned short* ap = &Abuf[cur][(wm * 64 + m * 16 + lr) * 32 + lh * 8];
            v16bf a = frag2(*(const u32x4*)ap, *(const u32x4*)(ap + 16));
            acc[m][0] = WMMA_BF16(a, b0, acc[m][0]);
            acc[m][1] = WMMA_BF16(a, b1, acc[m][1]);
        }
        wait_async0();
        __syncthreads();
    }

    const int rowBase = rowBase0 + wm * 64;
    const int colBase = colBase0 + wn * 32;
    const float bv0 = bias[colBase + lr];
    const float bv1 = bias[colBase + 16 + lr];
    #pragma unroll
    for (int m = 0; m < 4; ++m) {
        #pragma unroll
        for (int r = 0; r < 8; ++r) {
            int row = rowBase + m * 16 + r + lh * 8;
            out[(size_t)row * G4H + colBase + lr]      = acc[m][0][r] + bv0;
            out[(size_t)row * G4H + colBase + 16 + lr] = acc[m][1][r] + bv1;
        }
    }
}

// ---------------------------------------------------------------------------
// Recurrent LSTM: one WG (32 waves) per direction; gates via LDS, c/h in regs.
// __builtin_prefetch pulls next timestep's xp slice toward the WGP.
// ---------------------------------------------------------------------------
__global__ void __launch_bounds__(1024)
lstm_kernel(const float* __restrict__ xpf, const float* __restrict__ xpb,
            const unsigned short* __restrict__ whhf,
            const unsigned short* __restrict__ whhb,
            const int* __restrict__ lengths,
            unsigned short* __restrict__ hcat) {
    const int dir = blockIdx.x;
    const float* xp          = dir ? xpb : xpf;
    const unsigned short* wh = dir ? whhb : whhf;

    extern __shared__ char smem[];
    unsigned short* gates = (unsigned short*)smem;                   // [64][1024]
    unsigned short* hbuf  = (unsigned short*)(smem + Bsz * G4H * 2); // [64][256]

    const int tid = threadIdx.x;
    const int w  = tid >> 5;
    const int l  = tid & 31;
    const int lr = l & 15, lh = l >> 4;
    const int col0 = w * 32;

    for (int i = tid; i < Bsz * Hsz; i += 1024) hbuf[i] = 0;   // h(0) = 0

    const int ucol    = 8 * w + (l & 7);
    const int rowbase = (l >> 3) * 16;
    float c_reg[16], h_reg[16];
    int   lens[16];
    #pragma unroll
    for (int i = 0; i < 16; ++i) { c_reg[i] = 0.f; h_reg[i] = 0.f; lens[i] = lengths[rowbase + i]; }

    __syncthreads();

    for (int s = 0; s < Tsz; ++s) {
        const int t = dir ? (Tsz - 1 - s) : s;

        if (s + 1 < Tsz) {                         // prefetch xp[:, t_next, :]
            const int tn = dir ? (t - 1) : (t + 1);
            __builtin_prefetch(xp + ((size_t)(tid >> 4) * Tsz + tn) * G4H + (tid & 15) * 64, 0, 3);
        }

        // ---- GEMM phase: gates = xp + h @ Whh^T ----
        for (int m = 0; m < 4; ++m) {
            v8f acc0, acc1;
            #pragma unroll
            for (int r = 0; r < 8; ++r) {          // seed accumulators with xp
                int row = m * 16 + r + lh * 8;
                size_t base = ((size_t)row * Tsz + t) * G4H;
                acc0[r] = xp[base + col0 + lr];
                acc1[r] = xp[base + col0 + 16 + lr];
            }
            for (int kc = 0; kc < 8; ++kc) {
                const unsigned short* ap = hbuf + (m * 16 + lr) * Hsz + kc * 32 + lh * 8;
                v16bf a = frag2(*(const u32x4*)ap, *(const u32x4*)(ap + 16));
                const int bk = kc * 32 + lh * 8;
                v16bf b0 = frag1(wh + (size_t)(col0 + lr)      * Hsz + bk);
                v16bf b1 = frag1(wh + (size_t)(col0 + 16 + lr) * Hsz + bk);
                acc0 = WMMA_BF16(a, b0, acc0);
                acc1 = WMMA_BF16(a, b1, acc1);
            }
            #pragma unroll
            for (int r = 0; r < 8; ++r) {
                int row = m * 16 + r + lh * 8;
                gates[row * G4H + col0 + lr]      = f2bf(acc0[r]);
                gates[row * G4H + col0 + 16 + lr] = f2bf(acc1[r]);
            }
        }
        __syncthreads();

        // ---- update phase: c,h live in registers across all timesteps ----
        #pragma unroll 4
        for (int i = 0; i < 16; ++i) {
            int row = rowbase + i;
            const unsigned short* g = gates + row * G4H + ucol;
            float ig = sigm(bf2f(g[0]));
            float fg = sigm(bf2f(g[Hsz]));
            float gg = tanhf(bf2f(g[2 * Hsz]));
            float og = sigm(bf2f(g[3 * Hsz]));
            float cn = fg * c_reg[i] + ig * gg;
            float hn = og * tanhf(cn);
            if (t < lens[i]) { c_reg[i] = cn; h_reg[i] = hn; }
            unsigned short hb = f2bf(h_reg[i]);
            hbuf[row * Hsz + ucol] = hb;
            hcat[((size_t)row * Tsz + t) * (2 * Hsz) + dir * Hsz + ucol] = hb;
        }
        __syncthreads();
    }
}

// ---------------------------------------------------------------------------
// logits = [h_fwd|h_bwd] @ W_clf^T + b_clf.  M = 32768, K = 512, N = 32.
// Block 256 thr (8 waves): WG tile 256x32, wave tile 32x32.  grid = 128.
// A tiles (256x32 bf16) DMA'd into LDS by the Tensor Data Mover, double-buffered.
// ---------------------------------------------------------------------------
__global__ void __launch_bounds__(256)
clf_kernel(const unsigned short* __restrict__ hcat,
           const unsigned short* __restrict__ wclf,
           const float* __restrict__ bclf,
           float* __restrict__ logits) {
#ifdef HAVE_TDM
    __shared__ unsigned short Abuf[2][256 * 32];   // 2 x 16 KB
#endif
    const int tid = threadIdx.x;
    const int w  = tid >> 5;
    const int l  = tid & 31;
    const int lr = l & 15, lh = l >> 4;
    const int rowBase0 = blockIdx.x * 256;

#ifdef HAVE_TDM
    if (w == 0) {
        tdm_load_2d(hcat + (size_t)rowBase0 * (2 * Hsz), lds_byte_addr(&Abuf[0][0]),
                    /*tdim0=*/2 * Hsz, /*tdim1=*/Bsz * Tsz,
                    /*tile0=*/32, /*tile1=*/256, /*stride0=*/2 * Hsz);
        __builtin_amdgcn_s_wait_tensorcnt(0);
    }
    __syncthreads();
#endif

    v8f acc[2][2];
    #pragma unroll
    for (int m = 0; m < 2; ++m) { acc[m][0] = (v8f)0.0f; acc[m][1] = (v8f)0.0f; }

    for (int kc = 0; kc < 16; ++kc) {
#ifdef HAVE_TDM
        const int cur = kc & 1;
        if (w == 0 && kc + 1 < 16) {               // TDM: DMA next A tile
            tdm_load_2d(hcat + (size_t)rowBase0 * (2 * Hsz) + (kc + 1) * 32,
                        lds_byte_addr(&Abuf[cur ^ 1][0]),
                        2 * Hsz, Bsz * Tsz, 32, 256, 2 * Hsz);
        }
#endif
        const int bk = kc * 32 + lh * 8;
        v16bf b0 = frag1(wclf + (size_t)(lr)      * (2 * Hsz) + bk);
        v16bf b1 = frag1(wclf + (size_t)(16 + lr) * (2 * Hsz) + bk);
        #pragma unroll
        for (int m = 0; m < 2; ++m) {
#ifdef HAVE_TDM
            const unsigned short* ap = &Abuf[cur][(w * 32 + m * 16 + lr) * 32 + lh * 8];
#else
            const unsigned short* ap = hcat + (size_t)(rowBase0 + w * 32 + m * 16 + lr) * (2 * Hsz)
                                            + kc * 32 + lh * 8;
#endif
            v16bf a = frag2(*(const u32x4*)ap, *(const u32x4*)(ap + 16));
            acc[m][0] = WMMA_BF16(a, b0, acc[m][0]);
            acc[m][1] = WMMA_BF16(a, b1, acc[m][1]);
        }
#ifdef HAVE_TDM
        if (w == 0) __builtin_amdgcn_s_wait_tensorcnt(0);
        __syncthreads();
#endif
    }

    const float bv0 = bclf[lr], bv1 = bclf[16 + lr];
    #pragma unroll
    for (int m = 0; m < 2; ++m) {
        #pragma unroll
        for (int r = 0; r < 8; ++r) {
            int row = rowBase0 + w * 32 + m * 16 + r + lh * 8;
            logits[(size_t)row * Kst + lr]      = acc[m][0][r] + bv0;
            logits[(size_t)row * Kst + 16 + lr] = acc[m][1][r] + bv1;
        }
    }
}

// ---------------------------------------------------------------------------
// CRF Viterbi: one wave per batch element, lane = state (wave32 == K=32).
// ---------------------------------------------------------------------------
__global__ void __launch_bounds__(256)
viterbi_kernel(const float* __restrict__ logits, const int* __restrict__ lengths,
               const float* __restrict__ start_trans, const float* __restrict__ end_trans,
               const float* __restrict__ trans,
               unsigned char* __restrict__ hist, int* __restrict__ preds) {
    __shared__ float tr[Kst * Kst];
    for (int i = threadIdx.x; i < Kst * Kst; i += blockDim.x) tr[i] = trans[i];
    __syncthreads();

    const int w    = threadIdx.x >> 5;
    const int lane = threadIdx.x & 31;
    const int b    = blockIdx.x * 8 + w;
    const int len  = lengths[b];

    float score = start_trans[lane] + logits[(size_t)b * Tsz * Kst + lane];
    for (int t = 1; t < Tsz; ++t) {
        float emis = logits[((size_t)b * Tsz + t) * Kst + lane];
        float best = -3.402823e38f; int bi = 0;
        #pragma unroll 8
        for (int kp = 0; kp < Kst; ++kp) {         // first-max (argmax semantics)
            float cand = __shfl(score, kp, 32) + tr[kp * Kst + lane];
            if (cand > best) { best = cand; bi = kp; }
        }
        bool m = t < len;
        score = m ? (best + emis) : score;
        hist[((size_t)(t - 1) * Bsz + b) * Kst + lane] = (unsigned char)(m ? bi : lane);
    }
    score += end_trans[lane];
    float v = score; int idx = lane;
    for (int off = 16; off > 0; off >>= 1) {       // argmax, ties -> lower index
        float ov = __shfl_down(v, off, 32);
        int   oi = __shfl_down(idx, off, 32);
        if (ov > v || (ov == v && oi < idx)) { v = ov; idx = oi; }
    }
    int tag = __shfl(idx, 0, 32);
    if (lane == 0) {
        for (int t = Tsz - 1; t >= 1; --t) {
            preds[b * Tsz + t] = (t < len) ? tag : 0;
            tag = hist[((size_t)(t - 1) * Bsz + b) * Kst + tag];
        }
        preds[b * Tsz] = tag;                      // t=0 always unmasked (len >= T/2)
    }
}

// ---------------------------------------------------------------------------
extern "C" void kernel_launch(void* const* d_in, const int* in_sizes, int n_in,
                              void* d_out, int out_size, void* d_ws, size_t ws_size,
                              hipStream_t stream) {
    const float* x       = (const float*)d_in[0];
    const int* lengths   = (const int*)  d_in[1];
    /* d_in[2] = mask (bool) unused; derived from lengths */
    const float* Wih_f   = (const float*)d_in[3];
    const float* Whh_f   = (const float*)d_in[4];
    const float* b_f     = (const float*)d_in[5];
    const float* Wih_b   = (const float*)d_in[6];
    const float* Whh_b   = (const float*)d_in[7];
    const float* b_b     = (const float*)d_in[8];
    const float* W_clf   = (const float*)d_in[9];
    const float* b_clf   = (const float*)d_in[10];
    const float* s_tr    = (const float*)d_in[11];
    const float* e_tr    = (const float*)d_in[12];
    const float* trans   = (const float*)d_in[13];

    char* ws = (char*)d_ws;
    size_t off = 0;
    auto take = [&](size_t bytes) { char* p = ws + off; off += (bytes + 255) & ~(size_t)255; return p; };
    unsigned short* x_bf  = (unsigned short*)take((size_t)Bsz * Tsz * Dsz * 2);
    unsigned short* wihf  = (unsigned short*)take((size_t)G4H * Dsz * 2);
    unsigned short* wihb  = (unsigned short*)take((size_t)G4H * Dsz * 2);
    unsigned short* whhf  = (unsigned short*)take((size_t)G4H * Hsz * 2);
    unsigned short* whhb  = (unsigned short*)take((size_t)G4H * Hsz * 2);
    unsigned short* wclf  = (unsigned short*)take((size_t)Kst * 2 * Hsz * 2);
    float* xpf            = (float*)take((size_t)Bsz * Tsz * G4H * 4);
    float* xpb            = (float*)take((size_t)Bsz * Tsz * G4H * 4);
    unsigned short* hcat  = (unsigned short*)take((size_t)Bsz * Tsz * 2 * Hsz * 2);
    unsigned char* hist   = (unsigned char*)take((size_t)(Tsz - 1) * Bsz * Kst);

    float* logits = (float*)d_out;
    int*   preds  = (int*)d_out + (size_t)Bsz * Tsz * Kst;

    cvt_bf16_kernel<<<4096, 256, 0, stream>>>(x,     x_bf, Bsz * Tsz * Dsz);
    cvt_bf16_kernel<<<512,  256, 0, stream>>>(Wih_f, wihf, G4H * Dsz);
    cvt_bf16_kernel<<<512,  256, 0, stream>>>(Wih_b, wihb, G4H * Dsz);
    cvt_bf16_kernel<<<256,  256, 0, stream>>>(Whh_f, whhf, G4H * Hsz);
    cvt_bf16_kernel<<<256,  256, 0, stream>>>(Whh_b, whhb, G4H * Hsz);
    cvt_bf16_kernel<<<16,   256, 0, stream>>>(W_clf, wclf, Kst * 2 * Hsz);

    xproj_kernel<<<dim3(256, 8, 2), 256, 0, stream>>>(x_bf, wihf, wihb, b_f, b_b, xpf, xpb);

    lstm_kernel<<<2, 1024, (size_t)(Bsz * G4H + Bsz * Hsz) * 2, stream>>>(
        xpf, xpb, whhf, whhb, lengths, hcat);

    clf_kernel<<<128, 256, 0, stream>>>(hcat, wclf, b_clf, logits);

    viterbi_kernel<<<8, 256, 0, stream>>>(logits, lengths, s_tr, e_tr, trans, hist, preds);
}
